// ThreeTaskResNetFPNWithMoEDecodersAndSoftmaxInputs_39359080301105
// MI455X (gfx1250) — compile-verified, matrix operand order checked by reading, not run on hardware
//
#include <hip/hip_runtime.h>
#include <stddef.h>

// ---------------------------------------------------------------------------
// CDNA5 (gfx1250) implementation: ResNet50+FPN+mask pooling+expert decoder.
// All convolutions (incl. 7x7 stem, 3x3, 1x1, and the MLP GEMMs) run through a
// single implicit-GEMM kernel built on v_wmma_f32_16x16x32_f16 (wave32).
// Block tile 128x128, 8 waves (4x2), each wave 32x64 => 8 WMMAs per K-step.
// ---------------------------------------------------------------------------

typedef __attribute__((ext_vector_type(16))) _Float16 v16h;
typedef __attribute__((ext_vector_type(8)))  _Float16 v8h;
typedef __attribute__((ext_vector_type(8)))  float    v8f;

union HFrag { v16h v; v8h h[2]; };

#define BM 128
#define BN 128
#define BK 32
#define LDA (BK + 8)   // +8 halfs pad -> 80B row stride, keeps 16B alignment

__constant__ int c_counts[3][6] = {
    {7, 8, 6, 5, 9, 4},      // verb
    {10, 12, 8, 14, 9, 11},  // target
    {20, 25, 18, 30, 22, 28} // verbtarg
};

// ---------------------------------------------------------------------------
// Implicit-GEMM conv:  Out[n,o,oh,ow] = sum_{c,r,s} W[o,c,r,s]*X[n,c,ih,iw]
// A (weights) rows are contiguous (OIHW => row length Cin*KH*KW).
// B (activations) materialized on the fly (im2col) into LDS as f16, with an
// incremental (c,r,s) decode (no division in the inner loop).
// Epilogue: y = acc*scale[o] + bias[o] (+ residual) (ReLU optional).
// ---------------------------------------------------------------------------
__global__ __launch_bounds__(256) void conv_wmma_kernel(
    const float* __restrict__ Wgt, const float* __restrict__ Xin,
    float* __restrict__ Out,
    const float* __restrict__ scale, const float* __restrict__ bias,
    const float* __restrict__ resid,
    int N, int Cin, int Hin, int Win,
    int Cout, int KH, int KW, int stride, int pad,
    int Hout, int Wout, int relu)
{
  __shared__ __align__(16) _Float16 As[BM][LDA];
  __shared__ __align__(16) _Float16 Bs[BN][LDA];

  const int tid  = threadIdx.x;
  const int lane = tid & 31;
  const int wave = tid >> 5;
  const int wm   = wave >> 1;  // 0..3  (32 rows each)
  const int wn   = wave & 1;   // 0..1  (64 cols each)

  const int HWo   = Hout * Wout;
  const int Ncols = N * HWo;
  const int KHW   = KH * KW;
  const int numK  = Cin * KHW;

  const int rowBase = blockIdx.y * BM;
  const int colBase = blockIdx.x * BN;

  // B-tile staging: each thread owns one column and 16 consecutive k.
  const int bcol_l = tid >> 1;
  const int bkh    = (tid & 1) * 16;
  const int bcol_g = colBase + bcol_l;
  int bn_ = 0, boh = 0, bow = 0, bvalid = 0;
  if (bcol_g < Ncols) {
    bvalid = 1;
    bn_ = bcol_g / HWo;
    int hw = bcol_g - bn_ * HWo;
    boh = hw / Wout;
    bow = hw - boh * Wout;
  }
  const size_t xin_nbase = (size_t)bn_ * Cin;
  const int ih0 = boh * stride - pad;   // + r
  const int iw0 = bow * stride - pad;   // + s

  // A-tile staging: each thread owns one row and 16 consecutive k.
  const int arow_l = tid >> 1;
  const int akh    = (tid & 1) * 16;
  const int arow_g = rowBase + arow_l;
  const float* wp  = (arow_g < Cout) ? (Wgt + (size_t)arow_g * numK) : nullptr;

  v8f acc[2][4] = {};

  for (int kb = 0; kb < numK; kb += BK) {
    // ---- load A tile (weights, contiguous in k) ----
    {
#pragma unroll 4
      for (int j = 0; j < 16; ++j) {
        int kg = kb + akh + j;
        float v = (wp && kg < numK) ? wp[kg] : 0.f;
        As[arow_l][akh + j] = (_Float16)v;
      }
      // prefetch next K-slice of this weight row (global_prefetch_b8)
      if (wp && kb + BK < numK)
        __builtin_prefetch(&wp[kb + BK + akh], 0, 1);
    }
    // ---- load B tile (im2col on the fly; incremental (c,r,s) decode) ----
    {
      int kg = kb + bkh;
      int c = 0, r = 0, s = 0;
      if (kg < numK) {        // initial decode (once per K-step)
        c = kg / KHW;
        int rs = kg - c * KHW;
        r = rs / KW;
        s = rs - r * KW;
      }
#pragma unroll 4
      for (int j = 0; j < 16; ++j) {
        float v = 0.f;
        if (bvalid && kg < numK) {
          int ih = ih0 + r;
          int iw = iw0 + s;
          if ((unsigned)ih < (unsigned)Hin && (unsigned)iw < (unsigned)Win)
            v = Xin[((xin_nbase + c) * Hin + ih) * Win + iw];
        }
        Bs[bcol_l][bkh + j] = (_Float16)v;
        ++kg;
        if (++s == KW) { s = 0; if (++r == KH) { r = 0; ++c; } }
      }
    }
    __syncthreads();

    // ---- fragments: 16-bit A 16x32 layout (ISA 7.12.2):
    // lanes 0-15: row m0+l, K {0..7} then {16..23}; lanes 16-31: K {8..15},{24..31}
    const int l16  = lane & 15;
    const int hsel = lane >> 4;
    const int am0  = wm * 32;
    const int bn0  = wn * 64;
    HFrag af[2], bf[4];
#pragma unroll
    for (int i = 0; i < 2; ++i) {
      int rr = am0 + i * 16 + l16;
      af[i].h[0] = *(const v8h*)&As[rr][hsel * 8];
      af[i].h[1] = *(const v8h*)&As[rr][16 + hsel * 8];
    }
#pragma unroll
    for (int j = 0; j < 4; ++j) {
      int cc = bn0 + j * 16 + l16;
      bf[j].h[0] = *(const v8h*)&Bs[cc][hsel * 8];
      bf[j].h[1] = *(const v8h*)&Bs[cc][16 + hsel * 8];
    }
#pragma unroll
    for (int i = 0; i < 2; ++i)
#pragma unroll
      for (int j = 0; j < 4; ++j)
        acc[i][j] = __builtin_amdgcn_wmma_f32_16x16x32_f16(
            false, af[i].v, false, bf[j].v, (short)0, acc[i][j], false, false);

    __syncthreads();
  }

  // ---- epilogue: C/D layout: VGPR j, lane -> (m = j + (lane>=16)*8, n = lane&15)
  const int l16  = lane & 15;
  const int hsel = lane >> 4;
#pragma unroll
  for (int i = 0; i < 2; ++i) {
#pragma unroll
    for (int jn = 0; jn < 4; ++jn) {
      int oBase = rowBase + wm * 32 + i * 16 + hsel * 8;
      int col   = colBase + wn * 64 + jn * 16 + l16;
      if (col < Ncols) {
        int n  = col / HWo;
        int hw = col - n * HWo;
        size_t outb = (size_t)n * Cout * HWo + hw;
#pragma unroll
        for (int j = 0; j < 8; ++j) {
          int o = oBase + j;
          if (o < Cout) {
            float v  = acc[i][jn][j];
            float sc = scale ? scale[o] : 1.f;
            float bi = bias ? bias[o] : 0.f;
            v = v * sc + bi;
            size_t addr = outb + (size_t)o * HWo;
            if (resid) v += resid[addr];
            if (relu)  v = fmaxf(v, 0.f);
            Out[addr] = v;
          }
        }
      }
    }
  }
}

// ---------------------------------------------------------------------------
// Small elementwise / reduction kernels
// ---------------------------------------------------------------------------
__global__ void concat_img_mask_kernel(const float* __restrict__ img,
                                       const float* __restrict__ mask,
                                       float* __restrict__ X, int HW)
{
  int idx = blockIdx.x * blockDim.x + threadIdx.x;
  int total = 16 * 4 * HW;
  if (idx >= total) return;
  int p = idx % HW;
  int t = idx / HW;
  int c = t % 4;
  int n = t / 4;
  X[idx] = (c < 3) ? img[((size_t)n * 3 + c) * HW + p] : mask[(size_t)n * HW + p];
}

__global__ void maxpool3x3s2_kernel(const float* __restrict__ in, float* __restrict__ out,
                                    int C, int Hin, int Win, int Hout, int Wout)
{
  int idx = blockIdx.x * blockDim.x + threadIdx.x;
  int total = 16 * C * Hout * Wout;
  if (idx >= total) return;
  int ow = idx % Wout; int t = idx / Wout;
  int oh = t % Hout;   t /= Hout;
  int c  = t % C;      int n = t / C;
  float m = -__builtin_huge_valf();
  const float* base = in + ((size_t)n * C + c) * Hin * Win;
#pragma unroll
  for (int r = 0; r < 3; ++r) {
#pragma unroll
    for (int s = 0; s < 3; ++s) {
      int ih = oh * 2 - 1 + r, iw = ow * 2 - 1 + s;
      if ((unsigned)ih < (unsigned)Hin && (unsigned)iw < (unsigned)Win)
        m = fmaxf(m, base[ih * Win + iw]);
    }
  }
  out[idx] = m;
}

__global__ void upsample_add_kernel(float* __restrict__ lo, const float* __restrict__ hi,
                                    int C, int H, int W)
{
  int idx = blockIdx.x * blockDim.x + threadIdx.x;
  int total = 16 * C * H * W;
  if (idx >= total) return;
  int w = idx % W; int t = idx / W;
  int h = t % H;   t /= H;
  int c = t % C;   int n = t / C;
  lo[idx] += hi[(((size_t)n * C + c) * (H / 2) + (h >> 1)) * (W / 2) + (w >> 1)];
}

__global__ void pool_mask_kernel(const float* __restrict__ mask, float* __restrict__ sm,
                                 int Ho, int Wo, int S)
{
  int idx = blockIdx.x * blockDim.x + threadIdx.x;
  int total = 16 * Ho * Wo;
  if (idx >= total) return;
  int ow = idx % Wo; int t = idx / Wo;
  int oh = t % Ho;   int n = t / Ho;
  float s = 0.f;
  const float* base = mask + (size_t)n * 224 * 224;
  for (int r = 0; r < S; ++r)
    for (int c = 0; c < S; ++c)
      s += base[(oh * S + r) * 224 + (ow * S + c)];
  sm[idx] = s / (float)(S * S);
}

// inst[b, base+c] = mean_hw(f*sm) ; bg[b, base+c] = mean_hw(f*(1-sm))
__global__ __launch_bounds__(128) void masked_pool_kernel(
    const float* __restrict__ f, const float* __restrict__ sm,
    float* __restrict__ aggi, float* __restrict__ aggb, int C, int HW, int base)
{
  int c = blockIdx.x, b = blockIdx.y, t = threadIdx.x;
  const float* fp = f + ((size_t)b * C + c) * HW;
  const float* mp = sm + (size_t)b * HW;
  float si = 0.f, sb = 0.f;
  for (int i = t; i < HW; i += 128) {
    float fv = fp[i], mv = mp[i];
    si += fv * mv;
    sb += fv * (1.f - mv);
  }
  __shared__ float r1[128], r2[128];
  r1[t] = si; r2[t] = sb;
  __syncthreads();
  for (int s = 64; s > 0; s >>= 1) {
    if (t < s) { r1[t] += r1[t + s]; r2[t] += r2[t + s]; }
    __syncthreads();
  }
  if (t == 0) {
    aggi[(size_t)b * 1024 + base + c] = r1[0] / (float)HW;
    aggb[(size_t)b * 1024 + base + c] = r2[0] / (float)HW;
  }
}

__global__ void concat_cls_kernel(const float* __restrict__ aggi, const float* __restrict__ emb,
                                  const int* __restrict__ iid, float* __restrict__ cat)
{
  int idx = blockIdx.x * blockDim.x + threadIdx.x;
  if (idx >= 16 * 1088) return;
  int j = idx % 1088, b = idx / 1088;
  cat[idx] = (j < 1024) ? aggi[(size_t)b * 1024 + j]
                        : emb[(size_t)iid[b] * 64 + (j - 1024)];
}

__global__ __launch_bounds__(256) void l2norm_kernel(float* __restrict__ v)
{
  int b = blockIdx.x, t = threadIdx.x;
  float x = v[(size_t)b * 256 + t];
  __shared__ float red[256];
  red[t] = x * x;
  __syncthreads();
  for (int s = 128; s > 0; s >>= 1) {
    if (t < s) red[t] += red[t + s];
    __syncthreads();
  }
  float nrm = sqrtf(red[0]);
  v[(size_t)b * 256 + t] = x / (nrm + 1e-6f);
}

// out = LN(a + r) with per-sample expert gamma/beta (gathered by iid). D=256.
__global__ __launch_bounds__(256) void add_ln_kernel(
    const float* __restrict__ a, const float* __restrict__ r,
    const float* __restrict__ g, const float* __restrict__ bb,
    const int* __restrict__ iid, float* __restrict__ out)
{
  int b = blockIdx.x, t = threadIdx.x, e = iid[b];
  float v = a[(size_t)b * 256 + t] + r[(size_t)b * 256 + t];
  __shared__ float red[256];
  red[t] = v;
  __syncthreads();
  for (int s = 128; s > 0; s >>= 1) { if (t < s) red[t] += red[t + s]; __syncthreads(); }
  float mean = red[0] / 256.f;
  __syncthreads();
  float d = v - mean;
  red[t] = d * d;
  __syncthreads();
  for (int s = 128; s > 0; s >>= 1) { if (t < s) red[t] += red[t + s]; __syncthreads(); }
  float var = red[0] / 256.f;
  out[(size_t)b * 256 + t] =
      d * rsqrtf(var + 1e-5f) * g[(size_t)e * 256 + t] + bb[(size_t)e * 256 + t];
}

// Per-sample expert GEMV: out[b,o] = W[iid[b],o,:] . x[b,:] + bias[iid[b],o]
// task >= 0 applies the -inf padding mask from c_counts[task][iid[b]].
__global__ __launch_bounds__(256) void expert_gemv_kernel(
    const float* __restrict__ Wt, const float* __restrict__ bias,
    const float* __restrict__ x, const int* __restrict__ iid,
    float* __restrict__ out, int O, int D, int relu, int task)
{
  __shared__ float xs[2048];
  int b = blockIdx.y, e = iid[b];
  for (int i = threadIdx.x; i < D; i += 256) xs[i] = x[(size_t)b * D + i];
  __syncthreads();
  int o = blockIdx.x * 256 + threadIdx.x;
  if (o >= O) return;
  const float* wp = Wt + ((size_t)e * O + o) * D;
  float acc = bias ? bias[(size_t)e * O + o] : 0.f;
  for (int d = 0; d < D; ++d) acc += wp[d] * xs[d];
  if (relu) acc = fmaxf(acc, 0.f);
  if (task >= 0 && o >= c_counts[task][e]) acc = -__builtin_huge_valf();
  out[(size_t)b * O + o] = acc;
}

// ---------------------------------------------------------------------------
// Host orchestration
// ---------------------------------------------------------------------------
struct Blk { const float *w1,*s1,*b1,*w2,*s2,*b2,*w3,*s3,*b3,*dw,*ds,*db; };

extern "C" void kernel_launch(void* const* d_in, const int* in_sizes, int n_in,
                              void* d_out, int out_size, void* d_ws, size_t ws_size,
                              hipStream_t stream)
{
  (void)in_sizes; (void)out_size; (void)ws_size;

  const float* img  = (const float*)d_in[0];
  const float* mask = (const float*)d_in[1];
  const int*   iid  = (const int*)d_in[2];

  // Params: one device pointer per pytree leaf (sorted-key jax flatten order),
  // falling back to a single concatenated buffer if only 4 inputs arrive.
  bool leafMode = (n_in > 4);
  int pi = 3;
  const float* flat = leafMode ? nullptr : (const float*)d_in[3];
  size_t foff = 0;
  auto F = [&](size_t count) -> const float* {
    if (leafMode) return (const float*)d_in[pi++];
    const float* p = flat + foff; foff += count; return p;
  };

  // --- sorted top-level keys: bg_mlp, bn1_b, bn1_s, conv1_w, dec, emb,
  //     fc_target, fc_verb, fc_verbtarg, fpn_lat, fpn_out, inst_mlp, layer1..4
  const float *bg_w1 = F(128*1024), *bg_b1 = F(128), *bg_w2 = F(256*128), *bg_b2 = F(256);
  const float *bn1_b = F(64), *bn1_s = F(64), *conv1_w = F(64*4*49);
  // dec (sorted): ca_bo,ca_bv,ca_wo,ca_wv,ff_b1,ff_b2,ff_w1,ff_w2,ln1b,ln1g,...,sa_wv
  const float *ca_bo = F(6*256), *ca_bv = F(6*256), *ca_wo = F(6*256*256), *ca_wv = F(6*256*256);
  const float *ff_b1 = F(6*2048), *ff_b2 = F(6*256), *ff_w1 = F((size_t)6*2048*256), *ff_w2 = F((size_t)6*256*2048);
  const float *ln1b = F(6*256), *ln1g = F(6*256), *ln2b = F(6*256), *ln2g = F(6*256), *ln3b = F(6*256), *ln3g = F(6*256);
  const float *sa_bo = F(6*256), *sa_bv = F(6*256), *sa_wo = F(6*256*256), *sa_wv = F(6*256*256);
  const float *emb = F(6*64);
  const float *ft_b1  = F(6*256), *ft_b2  = F(6*14), *ft_w1  = F(6*256*256), *ft_w2  = F(6*14*256);
  const float *fv_b1  = F(6*256), *fv_b2  = F(6*9),  *fv_w1  = F(6*256*256), *fv_w2  = F(6*9*256);
  const float *fvt_b1 = F(6*256), *fvt_b2 = F(6*30), *fvt_w1 = F(6*256*256), *fvt_w2 = F(6*30*256);
  const float *lat_w[4], *lat_b[4];
  const int latc[4] = {256, 512, 1024, 2048};
  for (int i = 0; i < 4; ++i) { lat_w[i] = F((size_t)256*latc[i]); lat_b[i] = F(256); }
  const float *ow_[4], *ob_[4];
  for (int i = 0; i < 4; ++i) { ow_[i] = F((size_t)256*256*9); ob_[i] = F(256); }
  const float *im_w1 = F(128*1088), *im_b1 = F(128), *im_w2 = F(256*128), *im_b2 = F(256);

  auto parseBlk = [&](int cin, int cmid, int cout, bool down) -> Blk {
    Blk k{};  // sorted keys: b1,b2,b3,(db,ds,dw),s1,s2,s3,w1,w2,w3
    k.b1 = F(cmid); k.b2 = F(cmid); k.b3 = F(cout);
    if (down) { k.db = F(cout); k.ds = F(cout); k.dw = F((size_t)cout*cin); }
    k.s1 = F(cmid); k.s2 = F(cmid); k.s3 = F(cout);
    k.w1 = F((size_t)cmid*cin); k.w2 = F((size_t)cmid*cmid*9); k.w3 = F((size_t)cout*cmid);
    return k;
  };
  Blk L1b[3], L2b[4], L3b[6], L4b[3];
  L1b[0] = parseBlk(64, 64, 256, true);
  for (int j = 1; j < 3; ++j) L1b[j] = parseBlk(256, 64, 256, false);
  L2b[0] = parseBlk(256, 128, 512, true);
  for (int j = 1; j < 4; ++j) L2b[j] = parseBlk(512, 128, 512, false);
  L3b[0] = parseBlk(512, 256, 1024, true);
  for (int j = 1; j < 6; ++j) L3b[j] = parseBlk(1024, 256, 1024, false);
  L4b[0] = parseBlk(1024, 512, 2048, true);
  for (int j = 1; j < 3; ++j) L4b[j] = parseBlk(2048, 512, 2048, false);

  // --- workspace layout (floats) ---
  float* ws = (float*)d_ws;
  size_t off = 0;
  auto alloc = [&](size_t n) { float* p = ws + off; off += n; return p; };
  float* X    = alloc(3211264);    // [16,4,224,224]
  float* P0   = alloc(12845056);
  float* P1   = alloc(12845056);
  float* M0   = alloc(6422528);
  float* M1   = alloc(3211264);
  float* IDT  = alloc(12845056);
  float* POOL = alloc(3211264);    // [16,64,56,56]
  float* F1   = alloc(12845056);   // [16,256,56,56]
  float* F2   = alloc(6422528);    // [16,512,28,28]
  float* F3   = alloc(3211264);    // [16,1024,14,14]
  float* F4   = alloc(1605632);    // [16,2048,7,7]
  float* LAT1 = alloc(12845056);   // [16,256,56,56]
  float* LAT2 = alloc(3211264);
  float* LAT3 = alloc(802816);
  float* LAT4 = alloc(200704);
  float* FO   = alloc(12845056);   // reused per scale
  float* SM1  = alloc(16*3136);
  float* SM2  = alloc(16*784);
  float* SM3  = alloc(16*196);
  float* SM4  = alloc(16*49);
  float* AGGI = alloc(16*1024);
  float* AGGB = alloc(16*1024);
  float* CAT  = alloc(16*1088);
  float* INST = alloc(16*256);
  float* BG   = alloc(16*256);
  float* T0   = alloc(16*2048);
  float* T1   = alloc(16*2048);
  float* X1   = alloc(16*256);
  float* X2   = alloc(16*256);
  float* DECB = alloc(16*256);
  float* HH   = alloc(16*256);

  auto conv = [&](const float* Wg, const float* Xi, float* Ot,
                  const float* sc, const float* bi, const float* rs,
                  int N, int Cin, int Hin, int Win, int Cout,
                  int KH, int KW, int st, int pd, int relu) {
    int Hout = (Hin + 2*pd - KH) / st + 1;
    int Wout = (Win + 2*pd - KW) / st + 1;
    int Ncols = N * Hout * Wout;
    dim3 grid((Ncols + BN - 1) / BN, (Cout + BM - 1) / BM);
    conv_wmma_kernel<<<grid, dim3(256), 0, stream>>>(
        Wg, Xi, Ot, sc, bi, rs, N, Cin, Hin, Win, Cout, KH, KW, st, pd, Hout, Wout, relu);
  };

  // ---- stem ----
  {
    int tot = 16 * 4 * 224 * 224;
    concat_img_mask_kernel<<<(tot + 255) / 256, 256, 0, stream>>>(img, mask, X, 224 * 224);
  }
  conv(conv1_w, X, P0, bn1_s, bn1_b, nullptr, 16, 4, 224, 224, 64, 7, 7, 2, 3, 1);
  {
    int tot = 16 * 64 * 56 * 56;
    maxpool3x3s2_kernel<<<(tot + 255) / 256, 256, 0, stream>>>(P0, POOL, 64, 112, 112, 56, 56);
  }

  // ---- ResNet layers ----
  auto runLayer = [&](Blk* blks, int nblk, const float* xin, float* fout,
                      int cin, int cmid, int cout, int Hin, int st) {
    const float* x = xin;
    int Hi = Hin;
    for (int j = 0; j < nblk; ++j) {
      int s  = (j == 0) ? st : 1;
      int Ho = Hi / s;
      float* outp = (j == nblk - 1) ? fout : ((x == P0) ? P1 : P0);
      const float* idt;
      int ci = (j == 0) ? cin : cout;
      if (j == 0) {
        conv(blks[0].dw, x, IDT, blks[0].ds, blks[0].db, nullptr,
             16, ci, Hi, Hi, cout, 1, 1, s, 0, 0);
        idt = IDT;
      } else {
        idt = x;
      }
      conv(blks[j].w1, x,  M0, blks[j].s1, blks[j].b1, nullptr, 16, ci,   Hi, Hi, cmid, 1, 1, 1, 0, 1);
      conv(blks[j].w2, M0, M1, blks[j].s2, blks[j].b2, nullptr, 16, cmid, Hi, Hi, cmid, 3, 3, s, 1, 1);
      conv(blks[j].w3, M1, outp, blks[j].s3, blks[j].b3, idt,   16, cmid, Ho, Ho, cout, 1, 1, 1, 0, 1);
      x = outp;
      Hi = Ho;
    }
  };
  runLayer(L1b, 3, POOL, F1, 64,   64,  256,  56, 1);
  runLayer(L2b, 4, F1,   F2, 256,  128, 512,  56, 2);
  runLayer(L3b, 6, F2,   F3, 512,  256, 1024, 28, 2);
  runLayer(L4b, 3, F3,   F4, 1024, 512, 2048, 14, 2);

  // ---- FPN ----
  conv(lat_w[0], F1, LAT1, nullptr, lat_b[0], nullptr, 16, 256,  56, 56, 256, 1, 1, 1, 0, 0);
  conv(lat_w[1], F2, LAT2, nullptr, lat_b[1], nullptr, 16, 512,  28, 28, 256, 1, 1, 1, 0, 0);
  conv(lat_w[2], F3, LAT3, nullptr, lat_b[2], nullptr, 16, 1024, 14, 14, 256, 1, 1, 1, 0, 0);
  conv(lat_w[3], F4, LAT4, nullptr, lat_b[3], nullptr, 16, 2048, 7,  7,  256, 1, 1, 1, 0, 0);
  {
    int t3 = 16 * 256 * 14 * 14;
    upsample_add_kernel<<<(t3 + 255) / 256, 256, 0, stream>>>(LAT3, LAT4, 256, 14, 14);
    int t2 = 16 * 256 * 28 * 28;
    upsample_add_kernel<<<(t2 + 255) / 256, 256, 0, stream>>>(LAT2, LAT3, 256, 28, 28);
    int t1 = 16 * 256 * 56 * 56;
    upsample_add_kernel<<<(t1 + 255) / 256, 256, 0, stream>>>(LAT1, LAT2, 256, 56, 56);
  }

  // ---- pooled masks ----
  {
    int hos[4] = {56, 28, 14, 7};
    int ss[4]  = {4, 8, 16, 32};
    float* sms[4] = {SM1, SM2, SM3, SM4};
    for (int i = 0; i < 4; ++i) {
      int tot = 16 * hos[i] * hos[i];
      pool_mask_kernel<<<(tot + 255) / 256, 256, 0, stream>>>(mask, sms[i], hos[i], hos[i], ss[i]);
    }
  }

  // ---- FPN output convs + masked pooling (FO buffer reused) ----
  {
    int hos[4] = {56, 28, 14, 7};
    float* lats[4] = {LAT1, LAT2, LAT3, LAT4};
    float* sms[4]  = {SM1, SM2, SM3, SM4};
    for (int i = 0; i < 4; ++i) {
      conv(ow_[i], lats[i], FO, nullptr, ob_[i], nullptr, 16, 256, hos[i], hos[i], 256, 3, 3, 1, 1, 0);
      masked_pool_kernel<<<dim3(256, 16), 128, 0, stream>>>(
          FO, sms[i], AGGI, AGGB, 256, hos[i] * hos[i], i * 256);
    }
  }

  // ---- MLPs (run as 1x1 "conv" GEMMs with N=16, H=W=1) ----
  conv(bg_w1, AGGB, T0, nullptr, bg_b1, nullptr, 16, 1024, 1, 1, 128, 1, 1, 1, 0, 1);
  conv(bg_w2, T0,   BG, nullptr, bg_b2, nullptr, 16, 128,  1, 1, 256, 1, 1, 1, 0, 0);
  {
    int tot = 16 * 1088;
    concat_cls_kernel<<<(tot + 255) / 256, 256, 0, stream>>>(AGGI, emb, iid, CAT);
  }
  conv(im_w1, CAT, T0,   nullptr, im_b1, nullptr, 16, 1088, 1, 1, 128, 1, 1, 1, 0, 1);
  conv(im_w2, T0,  INST, nullptr, im_b2, nullptr, 16, 128,  1, 1, 256, 1, 1, 1, 0, 0);
  l2norm_kernel<<<16, 256, 0, stream>>>(INST);
  l2norm_kernel<<<16, 256, 0, stream>>>(BG);

  // ---- per-sample expert decoder ----
  auto gemv = [&](const float* W, const float* b, const float* x, float* o,
                  int O, int D, int relu, int task) {
    dim3 grid((O + 255) / 256, 16);
    expert_gemv_kernel<<<grid, 256, 0, stream>>>(W, b, x, iid, o, O, D, relu, task);
  };
  gemv(sa_wv, sa_bv, INST, T0, 256, 256, 0, -1);
  gemv(sa_wo, sa_bo, T0,   T1, 256, 256, 0, -1);
  add_ln_kernel<<<16, 256, 0, stream>>>(INST, T1, ln1g, ln1b, iid, X1);
  gemv(ca_wv, ca_bv, BG, T0, 256, 256, 0, -1);
  gemv(ca_wo, ca_bo, T0, T1, 256, 256, 0, -1);
  add_ln_kernel<<<16, 256, 0, stream>>>(X1, T1, ln2g, ln2b, iid, X2);
  gemv(ff_w1, ff_b1, X2, T0, 2048, 256, 1, -1);
  gemv(ff_w2, ff_b2, T0, T1, 256, 2048, 0, -1);
  add_ln_kernel<<<16, 256, 0, stream>>>(X2, T1, ln3g, ln3b, iid, DECB);

  // ---- heads (verb, target, verbtarg) -> d_out, with -inf padding mask ----
  float* OUTV  = (float*)d_out;        // [16,9]
  float* OUTT  = OUTV + 16 * 9;        // [16,14]
  float* OUTVT = OUTT + 16 * 14;       // [16,30]
  gemv(fv_w1,  fv_b1,  DECB, HH,    256, 256, 1, -1);
  gemv(fv_w2,  fv_b2,  HH,   OUTV,  9,   256, 0, 0);
  gemv(ft_w1,  ft_b1,  DECB, HH,    256, 256, 1, -1);
  gemv(ft_w2,  ft_b2,  HH,   OUTT,  14,  256, 0, 1);
  gemv(fvt_w1, fvt_b1, DECB, HH,    256, 256, 1, -1);
  gemv(fvt_w2, fvt_b2, HH,   OUTVT, 30,  256, 0, 2);
}
